// SelfMaskedAttention_8598524526661
// MI455X (gfx1250) — compile-verified, matrix operand order checked by reading, not run on hardware
//
#include <hip/hip_runtime.h>

// Problem constants
#define NB   2
#define SLEN 2048
#define EMB  1024
#define NH   16
#define HD   64

typedef _Float16 half8  __attribute__((ext_vector_type(8)));
typedef _Float16 half16 __attribute__((ext_vector_type(16)));
typedef float    float8 __attribute__((ext_vector_type(8)));

// Wave-uniform value -> SGPR (value must be uniform across the wave).
static __device__ inline int uni(int v) { return __builtin_amdgcn_readfirstlane(v); }

// Load a 16x32 f16 fragment (A-layout) from row-major memory with row stride ld.
// Lane l -> row (l&15); halves 0..7 = cols kb..kb+7, halves 8..15 = cols kb+16..kb+23,
// where kb = (l>=16) ? 8 : 0.  Two b128 loads per lane.
static __device__ inline half16 load_frag(const _Float16* __restrict__ base, int ld) {
    const int lane = threadIdx.x & 31;
    const _Float16* p = base + (size_t)(lane & 15) * ld + ((lane >> 4) << 3);
    half8 lo = *(const half8*)(p);
    half8 hi = *(const half8*)(p + 16);
    return __builtin_shufflevector(lo, hi, 0,1,2,3,4,5,6,7,8,9,10,11,12,13,14,15);
}

static __device__ inline float8 wmma16(half16 a, half16 b, float8 c) {
    // D = A(16x32 f16) x B(32x16 f16) + C(16x16 f32)
    return __builtin_amdgcn_wmma_f32_16x16x32_f16(false, a, false, b, (short)0, c,
                                                  false, false);
}

// ---------------------------------------------------------------- conversions
__global__ void f32_to_f16_kernel(const float* __restrict__ src,
                                  _Float16* __restrict__ dst, int n) {
    int i = blockIdx.x * blockDim.x + threadIdx.x;
    int stride = gridDim.x * blockDim.x;
    for (; i < n; i += stride) dst[i] = (_Float16)src[i];
}

// ---------------------------------------------------------------- QKV GEMM
// qkv = xh(4096x1024) @ wh(3072x1024)^T + bias, scattered into per-head
// Q[s][d], K[s][d], V^T[d][s] f16 buffers.  32x64 tile per wave (spill-free).
__global__ void __launch_bounds__(256)
qkv_gemm_kernel(const _Float16* __restrict__ xh,
                const _Float16* __restrict__ wh,
                const float*    __restrict__ bias,
                _Float16* __restrict__ qh,
                _Float16* __restrict__ kh,
                _Float16* __restrict__ vT) {
    const int wid = uni(blockIdx.x * (blockDim.x >> 5) + (threadIdx.x >> 5));
    const int m0 = (wid / 48) * 32;   // 128 M-tiles of 32 rows  (SGPR)
    const int n0 = (wid % 48) * 64;   // 48  N-tiles of 64 cols  (SGPR)

    float8 acc[2][4];
#pragma unroll
    for (int i = 0; i < 2; ++i)
#pragma unroll
        for (int j = 0; j < 4; ++j) acc[i][j] = {};

    for (int k0 = 0; k0 < EMB; k0 += 32) {
        half16 a[2], b[4];
#pragma unroll
        for (int i = 0; i < 2; ++i)
            a[i] = load_frag(xh + (size_t)(m0 + 16 * i) * EMB + k0, EMB);
#pragma unroll
        for (int j = 0; j < 4; ++j)
            b[j] = load_frag(wh + (size_t)(n0 + 16 * j) * EMB + k0, EMB);
#pragma unroll
        for (int i = 0; i < 2; ++i)
#pragma unroll
            for (int j = 0; j < 4; ++j)
                acc[i][j] = wmma16(a[i], b[j], acc[i][j]);
    }

    const int lane = threadIdx.x & 31;
    const int cn = lane & 15;          // C-frag: lane -> column
    const int rb = (lane >> 4) << 3;   // rows rb..rb+7 in regs 0..7
    const int bb = uni(m0 >> 11);      // whole 32-row tile is in one batch
    const int s0u = m0 & 2047;
#pragma unroll
    for (int j = 0; j < 4; ++j) {
        // Fragment-uniform routing: a 16-col fragment never straddles a
        // segment (Q/K/V) or a head (both boundaries are multiples of 16).
        const int colbase = n0 + j * 16;          // SGPR
        const int seg = colbase >> 10;            // 0=Q 1=K 2=V   (SGPR)
        const int e0 = colbase & 1023;
        const int h  = e0 >> 6;                   // head          (SGPR)
        const int d0 = e0 & 63;                   // d = d0 + cn
        const float bv = bias[colbase + cn];
#pragma unroll
        for (int i = 0; i < 2; ++i) {
            const int s0 = s0u + i * 16 + rb;     // first of 8 consecutive rows
            if (seg < 2) {
                _Float16* dst = (seg == 0 ? qh : kh) +
                    (((size_t)(bb * NH + h) * SLEN + s0) * HD) + d0 + cn;
#pragma unroll
                for (int r = 0; r < 8; ++r)
                    dst[(size_t)r * HD] = (_Float16)(acc[i][j][r] + bv);
            } else {
                _Float16* dst = vT +
                    (((size_t)(bb * NH + h) * HD + d0 + cn) * SLEN) + s0;
#pragma unroll
                for (int r = 0; r < 8; ++r)
                    dst[r] = (_Float16)(acc[i][j][r] + bv);
            }
        }
    }
}

// ---------------------------------------------------------------- attention
// One wave per (b,h, 16-query strip).  Scores computed transposed (S^T = K Q^T)
// so the score C-frags coincide with the A-frag layout of P for P@V.
__global__ void __launch_bounds__(256)
attn_kernel(const _Float16* __restrict__ qh,
            const _Float16* __restrict__ kh,
            const _Float16* __restrict__ vT,
            _Float16* __restrict__ ctxh) {
    const int wid = uni(blockIdx.x * (blockDim.x >> 5) + (threadIdx.x >> 5));
    const int bh = wid >> 7;            // 0..31  (b*16+h)   (SGPR)
    const int q0 = (wid & 127) << 4;    // query base        (SGPR)
    const int lane = threadIdx.x & 31;
    const int cn = lane & 15;
    const int rb = (lane >> 4) << 3;

    const _Float16* Qp = qh + (size_t)bh * SLEN * HD;   // [s][d]
    const _Float16* Kp = kh + (size_t)bh * SLEN * HD;   // [s][d]
    const _Float16* Vt = vT + (size_t)bh * HD * SLEN;   // [d][s]

    const half16 qa0 = load_frag(Qp + (size_t)q0 * HD, HD);       // d 0..31
    const half16 qa1 = load_frag(Qp + (size_t)q0 * HD + 32, HD);  // d 32..63

    float8 o[4];
#pragma unroll
    for (int j = 0; j < 4; ++j) o[j] = {};
    float rowsum = 0.0f;
    const int qcol = q0 + cn;   // this lane's query index (S^T layout: lane -> q)

    for (int j0 = 0; j0 < SLEN; j0 += 32) {
        const half16 k00 = load_frag(Kp + (size_t)j0 * HD, HD);
        const half16 k01 = load_frag(Kp + (size_t)j0 * HD + 32, HD);
        const half16 k10 = load_frag(Kp + (size_t)(j0 + 16) * HD, HD);
        const half16 k11 = load_frag(Kp + (size_t)(j0 + 16) * HD + 32, HD);
        float8 z = {};
        float8 st0 = wmma16(k00, qa0, z);  st0 = wmma16(k01, qa1, st0); // k j0..+15
        float8 st1 = wmma16(k10, qa0, z);  st1 = wmma16(k11, qa1, st1); // k +16..+31

        // mask (zero masked logits, per reference), scale by 1/sqrt(64),
        // exponentiate; result lands directly in P's A-fragment layout.
        half16 pa = {};
#pragma unroll
        for (int r = 0; r < 8; ++r) {
            const int k0i = j0 + rb + r;
            float l0 = (k0i <= qcol) ? st0[r] * 0.125f : 0.0f;
            float p0 = __expf(l0);
            rowsum += p0;
            pa[r] = (_Float16)p0;
            const int k1i = j0 + 16 + rb + r;
            float l1 = (k1i <= qcol) ? st1[r] * 0.125f : 0.0f;
            float p1 = __expf(l1);
            rowsum += p1;
            pa[8 + r] = (_Float16)p1;
        }
#pragma unroll
        for (int j = 0; j < 4; ++j) {
            const half16 vb = load_frag(Vt + (size_t)(j * 16) * SLEN + j0, SLEN);
            o[j] = wmma16(pa, vb, o[j]);    // ctx(16q x 16d) accumulate
        }
    }

    // combine per-q row sums: lanes l and l+16 hold complementary k-halves
    const float tot = rowsum + __shfl_xor(rowsum, 16, 32);

    // write ctx in the reference's permuted layout: col = d*H + h
    const int bb = bh >> 4, h = bh & 15;
    _Float16* cbase = ctxh + ((size_t)(bb * SLEN + q0 + rb)) * EMB + h;
#pragma unroll
    for (int j = 0; j < 4; ++j) {
        const int d = j * 16 + cn;   // ctx C-frag: lane -> d column
#pragma unroll
        for (int r = 0; r < 8; ++r) {
            const float rs = __shfl(tot, rb + r, 32);   // rowsum for this q row
            cbase[(size_t)r * EMB + d * NH] = (_Float16)(o[j][r] / rs);
        }
    }
}

// ---------------------------------------------------------------- out proj
__global__ void __launch_bounds__(256)
out_gemm_kernel(const _Float16* __restrict__ ch,
                const _Float16* __restrict__ wh,
                const float*    __restrict__ bias,
                float* __restrict__ out) {
    const int wid = uni(blockIdx.x * (blockDim.x >> 5) + (threadIdx.x >> 5));
    const int m0 = (wid >> 4) * 32;   // 128 M-tiles (SGPR)
    const int n0 = (wid & 15) * 64;   // 16  N-tiles (SGPR)

    float8 acc[2][4];
#pragma unroll
    for (int i = 0; i < 2; ++i)
#pragma unroll
        for (int j = 0; j < 4; ++j) acc[i][j] = {};

    for (int k0 = 0; k0 < EMB; k0 += 32) {
        half16 a[2], b[4];
#pragma unroll
        for (int i = 0; i < 2; ++i)
            a[i] = load_frag(ch + (size_t)(m0 + 16 * i) * EMB + k0, EMB);
#pragma unroll
        for (int j = 0; j < 4; ++j)
            b[j] = load_frag(wh + (size_t)(n0 + 16 * j) * EMB + k0, EMB);
#pragma unroll
        for (int i = 0; i < 2; ++i)
#pragma unroll
            for (int j = 0; j < 4; ++j)
                acc[i][j] = wmma16(a[i], b[j], acc[i][j]);
    }

    const int lane = threadIdx.x & 31;
    const int cn = lane & 15;
    const int rb = (lane >> 4) << 3;
#pragma unroll
    for (int j = 0; j < 4; ++j) {
        const int col = n0 + j * 16 + cn;
        const float bv = bias[col];
#pragma unroll
        for (int i = 0; i < 2; ++i) {
            float* obase = out + (size_t)(m0 + i * 16 + rb) * EMB + col;
#pragma unroll
            for (int r = 0; r < 8; ++r)
                obase[(size_t)r * EMB] = acc[i][j][r] + bv;
        }
    }
}

// ---------------------------------------------------------------- launcher
extern "C" void kernel_launch(void* const* d_in, const int* in_sizes, int n_in,
                              void* d_out, int out_size, void* d_ws, size_t ws_size,
                              hipStream_t stream) {
    const float* x     = (const float*)d_in[0];   // (2,2048,1024)
    const float* w_qkv = (const float*)d_in[1];   // (3072,1024)
    const float* b_qkv = (const float*)d_in[2];   // (3072,)
    const float* w_out = (const float*)d_in[3];   // (1024,1024)
    const float* b_out = (const float*)d_in[4];   // (1024,)

    char* ws = (char*)d_ws;
    size_t off = 0;
    _Float16* xh    = (_Float16*)(ws + off); off += (size_t)NB * SLEN * EMB * 2;      // 8 MB
    _Float16* wqkvh = (_Float16*)(ws + off); off += (size_t)3 * EMB * EMB * 2;        // 6 MB
    _Float16* wouth = (_Float16*)(ws + off); off += (size_t)EMB * EMB * 2;            // 2 MB
    _Float16* qh    = (_Float16*)(ws + off); off += (size_t)NB * NH * SLEN * HD * 2;  // 8 MB
    _Float16* kh    = (_Float16*)(ws + off); off += (size_t)NB * NH * SLEN * HD * 2;  // 8 MB
    _Float16* vT    = (_Float16*)(ws + off); off += (size_t)NB * NH * HD * SLEN * 2;  // 8 MB
    _Float16* ctxh  = (_Float16*)(ws + off); off += (size_t)NB * SLEN * EMB * 2;      // 8 MB
    if (off > ws_size) return;  // workspace too small (deterministic no-op)

    f32_to_f16_kernel<<<1024, 256, 0, stream>>>(x,     xh,    NB * SLEN * EMB);
    f32_to_f16_kernel<<<1024, 256, 0, stream>>>(w_qkv, wqkvh, 3 * EMB * EMB);
    f32_to_f16_kernel<<<512,  256, 0, stream>>>(w_out, wouth, EMB * EMB);

    // 128 x 48 = 6144 tiles, 8 waves per 256-thread block
    qkv_gemm_kernel<<<768, 256, 0, stream>>>(xh, wqkvh, b_qkv, qh, kh, vT);
    // 2*16*(2048/16) = 4096 waves
    attn_kernel<<<512, 256, 0, stream>>>(qh, kh, vT, ctxh);
    // 128 x 16 = 2048 tiles
    out_gemm_kernel<<<256, 256, 0, stream>>>(ctxh, wouth, b_out, (float*)d_out);
}
